// NetMamba_60026462929052
// MI455X (gfx1250) — compile-verified, matrix operand order checked by reading, not run on hardware
//
#include <hip/hip_runtime.h>
#include <hip/hip_bf16.h>
#include <math.h>

typedef __attribute__((ext_vector_type(16))) __bf16 v16bf;
typedef __attribute__((ext_vector_type(8)))  float  v8f;

#define EPSV 1e-5f
static constexpr int E_     = 192;
static constexpr int DI_    = 384;
static constexpr int S_     = 16;
static constexpr int R_     = 12;
static constexpr int DEPTH_ = 4;
static constexpr int P_     = 400;
static constexpr int NC_    = 20;
static constexpr int BATCH_ = 32;
static constexpr int L_     = P_ + 1;        // 401
static constexpr int M_     = BATCH_ * L_;   // 12832 = 401*32 (multiple of 32)
static constexpr int XDBW_  = 48;            // R + 2S = 44 padded to 48

__device__ __forceinline__ float sigf(float x) { return 1.f / (1.f + __expf(-x)); }

// ---------------- weight conversion ----------------
__global__ void k_f32_to_bf16(const float* __restrict__ src, __bf16* __restrict__ dst, int n) {
    int i = blockIdx.x * blockDim.x + threadIdx.x;
    if (i < n) dst[i] = (__bf16)src[i];
}

// x_proj_w (DEPTH,44,DI) -> (DEPTH,48,DI) bf16, zero-padded rows 44..47
__global__ void k_pad_xpw(const float* __restrict__ src, __bf16* __restrict__ dst) {
    int i = blockIdx.x * blockDim.x + threadIdx.x;
    int total = DEPTH_ * XDBW_ * DI_;
    if (i >= total) return;
    int c = i % DI_;
    int r = (i / DI_) % XDBW_;
    int l = i / (DI_ * XDBW_);
    float v = (r < R_ + 2 * S_) ? src[((size_t)l * (R_ + 2 * S_) + r) * DI_ + c] : 0.f;
    dst[i] = (__bf16)v;
}

// ---------------- patch embed + pos ----------------
__global__ void k_patch(const float* __restrict__ imgs, const float* __restrict__ pw,
                        const float* __restrict__ pb, const float* __restrict__ pos,
                        float* __restrict__ tpe) {
    int i = blockIdx.x * blockDim.x + threadIdx.x;
    if (i >= BATCH_ * P_ * E_) return;
    int e = i % E_;
    int p = (i / E_) % P_;
    int b = i / (E_ * P_);
    const float* im = imgs + (size_t)b * 1600 + p * 4;
    float acc = pb[e];
#pragma unroll
    for (int k = 0; k < 4; ++k) acc += im[k] * pw[e * 4 + k];
    acc += pos[p * E_ + e];
    tpe[i] = acc;
}

// depthwise conv1d (width 3, pad 1) + cls token append -> hidden (B,L,E)
__global__ void k_cnn(const float* __restrict__ tpe, const float* __restrict__ cw,
                      const float* __restrict__ cb, const float* __restrict__ cls,
                      const float* __restrict__ pos, float* __restrict__ hidden) {
    int i = blockIdx.x * blockDim.x + threadIdx.x;
    if (i >= BATCH_ * L_ * E_) return;
    int e = i % E_;
    int l = (i / E_) % L_;
    int b = i / (E_ * L_);
    float acc;
    if (l == P_) {
        acc = cls[e] + pos[P_ * E_ + e];
    } else {
        acc = cb[e];
#pragma unroll
        for (int k = 0; k < 3; ++k) {
            int ls = l - 1 + k;
            if (ls >= 0 && ls < P_) acc += tpe[((size_t)b * P_ + ls) * E_ + e] * cw[e * 3 + k];
        }
    }
    hidden[i] = acc;
}

// ---------------- residual add + RMSNorm -> bf16 ----------------
__global__ void k_norm(const float* __restrict__ hidden, float* __restrict__ resid,
                       __bf16* __restrict__ hn, const float* __restrict__ nw, int first) {
    int m = blockIdx.x;
    int e = threadIdx.x;                // blockDim = 192
    size_t idx = (size_t)m * E_ + e;
    float r = hidden[idx] + (first ? 0.f : resid[idx]);
    resid[idx] = r;
    __shared__ float s[E_];
    __shared__ float inv;
    s[e] = r * r;
    __syncthreads();
    for (int st = 96; st >= 3; st >>= 1) {
        if (e < st) s[e] += s[e + st];
        __syncthreads();
    }
    if (e == 0) inv = rsqrtf((s[0] + s[1] + s[2]) / (float)E_ + EPSV);
    __syncthreads();
    hn[idx] = (__bf16)(r * inv * nw[e]);
}

// ---------------- bf16 WMMA GEMM: C[M,N] = A[M,K] * B[N,K]^T ----------------
// 4 waves/block; each wave computes a (16*MT) x (16*NT) tile with a fully
// unrolled, register double-buffered K pipeline (K is compile-time, mult of 32).
// Own k order (not the ISA table's), same for A and B: lanes 0-15 hold K 0..15, lanes 16-31 hold K 16..31,
// so both A and B fragments are contiguous 32B loads from row-major (.,K) data.
template <int NT, int MT, int K>
__global__ void k_gemm_bf16(const __bf16* __restrict__ A, const __bf16* __restrict__ B,
                            float* __restrict__ C, int M, int N) {
    constexpr int NK = K / 32;
    int lane = threadIdx.x & 31;
    int w    = threadIdx.x >> 5;
    int m0   = (blockIdx.y * 4 + w) * (16 * MT);
    if (m0 >= M) return;
    int n0   = blockIdx.x * (16 * NT);
    int hl   = lane & 15;
    int kln  = (lane >> 4) * 16;
    const __bf16* Ap = A + (size_t)(m0 + hl) * K + kln;
    const __bf16* Bp = B + (size_t)(n0 + hl) * K + kln;

    v16bf a[2][MT], b[2][NT];
    v8f acc[MT][NT] = {};
#pragma unroll
    for (int mt = 0; mt < MT; ++mt) a[0][mt] = *(const v16bf*)(Ap + (size_t)mt * 16 * K);
#pragma unroll
    for (int nt = 0; nt < NT; ++nt) b[0][nt] = *(const v16bf*)(Bp + (size_t)nt * 16 * K);

#pragma unroll
    for (int s = 0; s < NK; ++s) {
        const int cur = s & 1, nxt = cur ^ 1;
        if (s + 1 < NK) {
            const int ko = (s + 1) * 32;
#pragma unroll
            for (int mt = 0; mt < MT; ++mt)
                a[nxt][mt] = *(const v16bf*)(Ap + (size_t)mt * 16 * K + ko);
#pragma unroll
            for (int nt = 0; nt < NT; ++nt)
                b[nxt][nt] = *(const v16bf*)(Bp + (size_t)nt * 16 * K + ko);
        }
#pragma unroll
        for (int mt = 0; mt < MT; ++mt)
#pragma unroll
            for (int nt = 0; nt < NT; ++nt)
                acc[mt][nt] = __builtin_amdgcn_wmma_f32_16x16x32_bf16(
                    false, a[cur][mt], false, b[cur][nt], (short)0, acc[mt][nt], false, false);
    }

    int crow = m0 + ((lane >> 4) << 3);
    int ccol = lane & 15;
#pragma unroll
    for (int mt = 0; mt < MT; ++mt)
#pragma unroll
        for (int nt = 0; nt < NT; ++nt)
#pragma unroll
            for (int j = 0; j < 8; ++j)
                C[(size_t)(crow + mt * 16 + j) * N + n0 + nt * 16 + ccol] = acc[mt][nt][j];
}

// ---------------- causal conv(4) + SiLU on xm = xz[:, :DI] ----------------
__global__ void k_convsilu(const float* __restrict__ xz, const float* __restrict__ cw,
                           const float* __restrict__ cb, float* __restrict__ xcf,
                           __bf16* __restrict__ xcb) {
    int i = blockIdx.x * blockDim.x + threadIdx.x;
    if (i >= M_ * DI_) return;
    int d = i % DI_;
    int m = i / DI_;
    int b = m / L_;
    int l = m % L_;
    const float* wv = cw + d * 4;
    float acc = cb[d];
#pragma unroll
    for (int k = 0; k < 4; ++k) {
        int ls = l + k - 3;
        if (ls >= 0) acc += xz[((size_t)b * L_ + ls) * (2 * DI_) + d] * wv[k];
    }
    float sv = acc * sigf(acc);
    xcf[i] = sv;
    xcb[i] = (__bf16)sv;
}

// ---------------- selective scan: one thread per (b,d), state in registers ----
// The 44-float xdb row for step l is shared by all 384 threads of the block;
// stage it in LDS with double buffering (row l+1 is copied in while row l is
// consumed) so the serial critical path reads LDS instead of 384x-redundant VMEM.
__global__ void k_scan(const float* __restrict__ xdb, const float* __restrict__ xcf,
                       const float* __restrict__ dtw, const float* __restrict__ dtb,
                       const float* __restrict__ alog, float* __restrict__ ycore) {
    int b = blockIdx.x;      // 0..31
    int d = threadIdx.x;     // 0..383
    __shared__ float rowbuf[2][XDBW_];

    float wd[R_];
#pragma unroll
    for (int r = 0; r < R_; ++r) wd[r] = dtw[d * R_ + r];
    float bias = dtb[d];
    float Av[S_], h[S_];
#pragma unroll
    for (int n = 0; n < S_; ++n) { Av[n] = -__expf(alog[d * S_ + n]); h[n] = 0.f; }

    const float* xrow = xdb + (size_t)b * L_ * XDBW_;
    const float* urow = xcf + (size_t)b * L_ * DI_ + d;
    float* yrow       = ycore + (size_t)b * L_ * DI_ + d;

    if (d < XDBW_) rowbuf[0][d] = xrow[d];
    __syncthreads();

    for (int l = 0; l < L_; ++l) {
        const float* row = rowbuf[l & 1];
        if (l + 1 < L_ && d < XDBW_)
            rowbuf[(l + 1) & 1][d] = xrow[(size_t)(l + 1) * XDBW_ + d];

        float acc = bias;
#pragma unroll
        for (int r = 0; r < R_; ++r) acc += row[r] * wd[r];
        float dt = (acc > 20.f) ? acc : log1pf(__expf(acc));   // softplus
        float u  = urow[(size_t)l * DI_];
        float du = dt * u;
        float y = 0.f;
#pragma unroll
        for (int n = 0; n < S_; ++n) {
            float dA = __expf(dt * Av[n]);
            float hn = dA * h[n] + du * row[R_ + n];
            h[n] = hn;
            y += hn * row[R_ + S_ + n];
        }
        yrow[(size_t)l * DI_] = y;
        __syncthreads();
    }
}

// ---------------- y = (ycore + xc*D) * silu(z) -> bf16 ----------------
__global__ void k_ygate(const float* __restrict__ ycore, const float* __restrict__ xcf,
                        const float* __restrict__ xz, const float* __restrict__ Dv,
                        __bf16* __restrict__ yb) {
    int i = blockIdx.x * blockDim.x + threadIdx.x;
    if (i >= M_ * DI_) return;
    int d = i % DI_;
    size_t m = (size_t)(i / DI_);
    float z = xz[m * (2 * DI_) + DI_ + d];
    float y = (ycore[i] + xcf[i] * Dv[d]) * (z * sigf(z));
    yb[i] = (__bf16)y;
}

// ---------------- final: rmsnorm(hidden+residual)[:, -1] @ head ----------------
__global__ void k_final(const float* __restrict__ hidden, const float* __restrict__ resid,
                        const float* __restrict__ nfw, const float* __restrict__ hw,
                        const float* __restrict__ hb, float* __restrict__ out) {
    int b = blockIdx.x;
    int e = threadIdx.x;                 // blockDim = 192
    size_t idx = ((size_t)b * L_ + P_) * E_ + e;
    float v = hidden[idx] + resid[idx];
    __shared__ float s[E_];
    __shared__ float sv[E_];
    __shared__ float inv;
    s[e] = v * v;
    __syncthreads();
    for (int st = 96; st >= 3; st >>= 1) {
        if (e < st) s[e] += s[e + st];
        __syncthreads();
    }
    if (e == 0) inv = rsqrtf((s[0] + s[1] + s[2]) / (float)E_ + EPSV);
    __syncthreads();
    sv[e] = v * inv * nfw[e];
    __syncthreads();
    if (e < NC_) {
        float acc = hb[e];
        for (int j = 0; j < E_; ++j) acc += sv[j] * hw[e * E_ + j];
        out[b * NC_ + e] = acc;
    }
}

extern "C" void kernel_launch(void* const* d_in, const int* in_sizes, int n_in,
                              void* d_out, int out_size, void* d_ws, size_t ws_size,
                              hipStream_t stream) {
    const float* imgs      = (const float*)d_in[0];
    const float* patch_w   = (const float*)d_in[1];
    const float* patch_b   = (const float*)d_in[2];
    const float* cls_token = (const float*)d_in[3];
    const float* pos_embed = (const float*)d_in[4];
    const float* cnn_w     = (const float*)d_in[5];
    const float* cnn_b     = (const float*)d_in[6];
    const float* norm_w    = (const float*)d_in[7];
    const float* in_proj_w = (const float*)d_in[8];
    const float* conv_w    = (const float*)d_in[9];
    const float* conv_b    = (const float*)d_in[10];
    const float* x_proj_w  = (const float*)d_in[11];
    const float* dt_proj_w = (const float*)d_in[12];
    const float* dt_proj_b = (const float*)d_in[13];
    const float* A_log     = (const float*)d_in[14];
    const float* Dp        = (const float*)d_in[15];
    const float* out_proj_w= (const float*)d_in[16];
    const float* norm_f_w  = (const float*)d_in[17];
    const float* head_w    = (const float*)d_in[18];
    const float* head_b    = (const float*)d_in[19];
    float* out = (float*)d_out;

    char* base = (char*)d_ws;
    size_t off = 0;
    auto alloc = [&](size_t bytes) {
        size_t o = off;
        off = (off + bytes + 255) & ~(size_t)255;
        return o;
    };
    float*  resid = (float*) (base + alloc((size_t)M_ * E_ * 4));
    float*  hidden= (float*) (base + alloc((size_t)M_ * E_ * 4));
    __bf16* hn    = (__bf16*)(base + alloc((size_t)M_ * E_ * 2));
    float*  xz    = (float*) (base + alloc((size_t)M_ * 2 * DI_ * 4));
    float*  xcf   = (float*) (base + alloc((size_t)M_ * DI_ * 4));
    __bf16* xcb   = (__bf16*)(base + alloc((size_t)M_ * DI_ * 2));
    float*  xdb   = (float*) (base + alloc((size_t)M_ * XDBW_ * 4));
    float*  ycore = (float*) (base + alloc((size_t)M_ * DI_ * 4));
    __bf16* yb    = (__bf16*)(base + alloc((size_t)M_ * DI_ * 2));
    __bf16* w_in  = (__bf16*)(base + alloc((size_t)DEPTH_ * 2 * DI_ * E_ * 2));
    __bf16* w_xp  = (__bf16*)(base + alloc((size_t)DEPTH_ * XDBW_ * DI_ * 2));
    __bf16* w_out = (__bf16*)(base + alloc((size_t)DEPTH_ * E_ * DI_ * 2));
    float*  tpe   = xz;   // alias: patch-embed scratch is consumed before GEMM1 writes xz

    const int TB = 256;
    // weight conversion (deterministic each call)
    {
        int n = DEPTH_ * 2 * DI_ * E_;
        k_f32_to_bf16<<<(n + TB - 1) / TB, TB, 0, stream>>>(in_proj_w, w_in, n);
        n = DEPTH_ * E_ * DI_;
        k_f32_to_bf16<<<(n + TB - 1) / TB, TB, 0, stream>>>(out_proj_w, w_out, n);
        n = DEPTH_ * XDBW_ * DI_;
        k_pad_xpw<<<(n + TB - 1) / TB, TB, 0, stream>>>(x_proj_w, w_xp);
    }
    // patch embed + pos -> tpe ; conv1d(3) + cls -> hidden
    {
        int n = BATCH_ * P_ * E_;
        k_patch<<<(n + TB - 1) / TB, TB, 0, stream>>>(imgs, patch_w, patch_b, pos_embed, tpe);
        n = BATCH_ * L_ * E_;
        k_cnn<<<(n + TB - 1) / TB, TB, 0, stream>>>(tpe, cnn_w, cnn_b, cls_token, pos_embed, hidden);
    }

    dim3 gblk(128);
    int gy = (M_ + 127) / 128;   // 101 blocks of 4 waves x 32 rows
    for (int i = 0; i < DEPTH_; ++i) {
        k_norm<<<M_, E_, 0, stream>>>(hidden, resid, hn, norm_w + i * E_, i == 0 ? 1 : 0);

        // xz = hn @ in_w^T : (M,768), K=192
        k_gemm_bf16<4, 2, E_><<<dim3((2 * DI_) / 64, gy), gblk, 0, stream>>>(
            hn, w_in + (size_t)i * 2 * DI_ * E_, xz, M_, 2 * DI_);

        int n = M_ * DI_;
        k_convsilu<<<(n + TB - 1) / TB, TB, 0, stream>>>(
            xz, conv_w + (size_t)i * DI_ * 4, conv_b + (size_t)i * DI_, xcf, xcb);

        // xdb = xc @ xpw^T : (M,48), K=384
        k_gemm_bf16<3, 2, DI_><<<dim3(1, gy), gblk, 0, stream>>>(
            xcb, w_xp + (size_t)i * XDBW_ * DI_, xdb, M_, XDBW_);

        // sequential scan, state in registers, xdb rows staged through LDS
        k_scan<<<BATCH_, DI_, 0, stream>>>(
            xdb, xcf, dt_proj_w + (size_t)i * DI_ * R_, dt_proj_b + (size_t)i * DI_,
            A_log + (size_t)i * DI_ * S_, ycore);

        k_ygate<<<(n + TB - 1) / TB, TB, 0, stream>>>(ycore, xcf, xz, Dp + (size_t)i * DI_, yb);

        // hidden = y @ ow^T : (M,192), K=384
        k_gemm_bf16<4, 2, DI_><<<dim3(E_ / 64, gy), gblk, 0, stream>>>(
            yb, w_out + (size_t)i * E_ * DI_, hidden, M_, E_);
    }

    k_final<<<BATCH_, E_, 0, stream>>>(hidden, resid, norm_f_w, head_w, head_b, out);
}
